// DI_net_73091753443562
// MI455X (gfx1250) — matrix operand order repaired and checked by reading, not compiled, not run on hardware
//
#include <hip/hip_runtime.h>
#include <hip/hip_bf16.h>

typedef _Float16 v16h __attribute__((ext_vector_type(16)));
typedef _Float16 v8h  __attribute__((ext_vector_type(8)));
typedef _Float16 v4h  __attribute__((ext_vector_type(4)));
typedef float    v8f  __attribute__((ext_vector_type(8)));

#define B_BATCH   256
#define M_DIM     1024
#define N_DIM     256
#define K_DIM     512
#define TM        128
#define TN        128
#define KC        32
#define NCHUNK    (K_DIM / KC)   // 16
#define LDS_STR   40             // halves per row (32 data + 8 pad): conflict-free, 16B-aligned
#define WEIGHT_ELEMS (256ULL * 1024ULL * 256ULL)

// ---------------------------------------------------------------------------
// Prep: w[b,o] = lrelu(x @ mw_W.T + mw_b)        (-> ws, f32 [256,512])
//       bias_out[b,o] = lrelu(x @ mb_W.T + mb_b) (-> d_out tail, [256,256])
//       leftH = f16(left)                        (-> ws, f16 [1024,512])
// ---------------------------------------------------------------------------
__global__ __launch_bounds__(256) void prep_kernel(
    const float* __restrict__ x,
    const float* __restrict__ mwW, const float* __restrict__ mwb,
    const float* __restrict__ mbW, const float* __restrict__ mbb,
    const float* __restrict__ left,
    float* __restrict__ wbuf, float* __restrict__ bias_out,
    _Float16* __restrict__ leftH)
{
    int idx = blockIdx.x * 256 + threadIdx.x;
    if (idx < B_BATCH * 512) {
        int b = idx >> 9, o = idx & 511;
        float s = mwb[o];
        #pragma unroll
        for (int k = 0; k < 24; ++k) s += x[b * 24 + k] * mwW[o * 24 + k];
        wbuf[idx] = (s > 0.0f) ? s : 0.01f * s;
    } else if (idx < B_BATCH * 512 + B_BATCH * 256) {
        int j = idx - B_BATCH * 512;
        int b = j >> 8, o = j & 255;
        float s = mbb[o];
        #pragma unroll
        for (int k = 0; k < 24; ++k) s += x[b * 24 + k] * mbW[o * 24 + k];
        bias_out[j] = (s > 0.0f) ? s : 0.01f * s;
    } else {
        int j = idx - (B_BATCH * 512 + B_BATCH * 256);   // 0 .. 131071
        if (j < (M_DIM * K_DIM) / 4) {
            const float4 v = *(const float4*)(left + (size_t)j * 4);
            v4h h = { (_Float16)v.x, (_Float16)v.y, (_Float16)v.z, (_Float16)v.w };
            *(v4h*)(leftH + (size_t)j * 4) = h;
        }
    }
}

// Fragment loaders (ISA 16-bit WMMA VGPR layouts, wave32)
__device__ __forceinline__ v16h load_fragA(const _Float16* p) {
    // A lane (m = lane&15): elems 0..7 = K[8h..8h+7], elems 8..15 = K[16+8h..]
    v8h x0 = *(const v8h*)(p);
    v8h x1 = *(const v8h*)(p + 16);
    return __builtin_shufflevector(x0, x1, 0,1,2,3,4,5,6,7,8,9,10,11,12,13,14,15);
}
__device__ __forceinline__ v16h load_fragB(const _Float16* p) {
    // B lane (n = lane&15): elems 0..7 = K[8h..8h+7], elems 8..15 = K[16+8h..]
    v8h x0 = *(const v8h*)(p);
    v8h x1 = *(const v8h*)(p + 16);
    return __builtin_shufflevector(x0, x1, 0,1,2,3,4,5,6,7,8,9,10,11,12,13,14,15);
}

// CDNA5 async global->LDS copy (16B per lane), tracked with ASYNCcnt
__device__ __forceinline__ void async_copy_b128(unsigned int ldsoff, unsigned long long gaddr) {
    asm volatile("global_load_async_to_lds_b128 %0, %1, off"
                 :: "v"(ldsoff), "v"(gaddr) : "memory");
}
__device__ __forceinline__ void wait_asynccnt0() {
    asm volatile("s_wait_asynccnt 0" ::: "memory");
}

// ---------------------------------------------------------------------------
// Main: out[b] = left @ (diag(w_b) @ right), f16 WMMA, f32 accumulate.
// grid = (2, 8, 256), block = 256 threads = 8 waves.
// A tile: async DMA (f16 ws copy -> LDS).  B tile: reg prefetch + scale + cvt.
// ---------------------------------------------------------------------------
__global__ __launch_bounds__(256) void modgemm_kernel(
    const _Float16* __restrict__ leftH, // [1024, 512] f16 row-major (ws)
    const float* __restrict__ right,    // [512, 256]  row-major
    const float* __restrict__ wbuf,     // [256, 512]
    float* __restrict__ out)            // [256, 1024, 256] flat
{
    __shared__ _Float16 sA[2][TM * LDS_STR];
    __shared__ _Float16 sB[2][TN * LDS_STR];

    const int nbase = blockIdx.x * TN;
    const int mbase = blockIdx.y * TM;
    const int b     = blockIdx.z;

    const int tid  = threadIdx.x;
    const int wave = tid >> 5;
    const int lane = tid & 31;
    const int lrow = lane & 15;
    const int half = lane >> 4;

    // B staging mapping (fixed trip counts -> no predication)
    const int bn  = tid & 127;         // B: n index
    const int bg0 = tid >> 7;          // B: s'-group start 0..1, step 2

    const float* wrow = wbuf + (size_t)b * K_DIM;

    v8f acc[8];
    #pragma unroll
    for (int i = 0; i < 8; ++i) acc[i] = (v8f){0,0,0,0,0,0,0,0};

    float bPre[4][4];   // B prefetch registers (right values)
    float wPre[4][4];   // w scalars for B scaling

    // ---- A tile: async DMA leftH -> LDS buffer p (2 x b128 per lane) ----
    auto stage_A_async = [&](int kk, int p) {
        const _Float16* gbase = leftH + (size_t)mbase * K_DIM + kk;
        #pragma unroll
        for (int j = 0; j < 2; ++j) {
            const int o   = tid * 2 + j;      // 0..511
            const int row = o >> 2;           // 0..127
            const int seg = (o & 3) * 8;      // halves
            unsigned int ldsoff =
                (unsigned int)(uintptr_t)(&sA[p][row * LDS_STR + seg]);
            unsigned long long ga =
                (unsigned long long)(uintptr_t)(gbase + (size_t)row * K_DIM + seg);
            async_copy_b128(ldsoff, ga);
        }
    };
    // ---- B tile: global -> registers for chunk kk ----
    auto stage_B_loads = [&](int kk) {
        #pragma unroll
        for (int it = 0; it < 4; ++it) {
            const int g = bg0 + it * 2;
            #pragma unroll
            for (int i = 0; i < 4; ++i) {
                const int s = g * 4 + i;
                bPre[it][i] = right[(size_t)(kk + s) * N_DIM + nbase + bn];
                wPre[it][i] = wrow[kk + s];
            }
        }
    };
    // ---- B tile: registers -> LDS buffer p (scale by w, f32 -> f16) ----
    auto stage_B_store = [&](int p) {
        #pragma unroll
        for (int it = 0; it < 4; ++it) {
            const int g = bg0 + it * 2;
            v4h h = { (_Float16)(wPre[it][0] * bPre[it][0]),
                      (_Float16)(wPre[it][1] * bPre[it][1]),
                      (_Float16)(wPre[it][2] * bPre[it][2]),
                      (_Float16)(wPre[it][3] * bPre[it][3]) };
            *(v4h*)(&sB[p][bn * LDS_STR + g * 4]) = h;
        }
    };
    // ---- 8 WMMAs from LDS buffer p, B-fragment software pipeline ----
    auto compute = [&](int p) {
        const _Float16* sAp = &sA[p][0];
        const _Float16* sBp = &sB[p][0];
        const v16h a = load_fragA(sAp + (wave * 16 + lrow) * LDS_STR + 8 * half);
        v16h bcur = load_fragB(sBp + lrow * LDS_STR + 8 * half);
        #pragma unroll
        for (int cb = 0; cb < 8; ++cb) {
            v16h bnxt = bcur;
            if (cb < 7)
                bnxt = load_fragB(sBp + ((cb + 1) * 16 + lrow) * LDS_STR + 8 * half);
            acc[cb] = __builtin_amdgcn_wmma_f32_16x16x32_f16(
                          false, a, false, bcur, (short)0, acc[cb], false, false);
            bcur = bnxt;
        }
    };

    // prologue: fill buffer 0 with chunk 0
    stage_A_async(0, 0);
    stage_B_loads(0);
    stage_B_store(0);
    wait_asynccnt0();
    __syncthreads();

    // steady state: compute chunk c from buf (c&1) while DMA/fetching chunk c+1
    for (int c = 0; c < NCHUNK - 1; ++c) {
        stage_A_async((c + 1) * KC, (c + 1) & 1);  // async DMA into other buffer
        stage_B_loads((c + 1) * KC);               // issue global loads early
        compute(c & 1);                            // WMMAs overlap load latency
        stage_B_store((c + 1) & 1);                // scale + cvt + ds_store
        wait_asynccnt0();
        __syncthreads();
    }
    compute((NCHUNK - 1) & 1);                     // last chunk

    // ---- store: C VGPR r -> row (8*half + r), col (lane&15) ----
    float* outb = out + (size_t)b * M_DIM * N_DIM;
    const int rowb = mbase + wave * 16 + half * 8;
    #pragma unroll
    for (int cb = 0; cb < 8; ++cb) {
        const int col = nbase + cb * 16 + lrow;
        #pragma unroll
        for (int r = 0; r < 8; ++r) {
            outb[(size_t)(rowb + r) * N_DIM + col] = acc[cb][r];
        }
    }
}

extern "C" void kernel_launch(void* const* d_in, const int* in_sizes, int n_in,
                              void* d_out, int out_size, void* d_ws, size_t ws_size,
                              hipStream_t stream) {
    const float* x    = (const float*)d_in[0];
    const float* mwW  = (const float*)d_in[1];
    const float* mwb  = (const float*)d_in[2];
    const float* mbW  = (const float*)d_in[3];
    const float* mbb  = (const float*)d_in[4];
    const float* left = (const float*)d_in[5];
    const float* right= (const float*)d_in[6];

    float*     out   = (float*)d_out;
    float*     wbuf  = (float*)d_ws;                               // 512 KB
    _Float16*  leftH = (_Float16*)((char*)d_ws + 512 * 1024);      // 1 MB

    // w (131072) + bias_out (65536) + leftH cvt (131072 float4 tasks)
    prep_kernel<<<1280, 256, 0, stream>>>(x, mwW, mwb, mbW, mbb, left,
                                          wbuf, out + WEIGHT_ELEMS, leftH);

    dim3 grid(N_DIM / TN, M_DIM / TM, B_BATCH);   // (2, 8, 256)
    modgemm_kernel<<<grid, 256, 0, stream>>>(leftH, right, wbuf, out);
}
